// MambaLinearProbe_87359634801440
// MI455X (gfx1250) — compile-verified, matrix operand order checked by reading, not run on hardware
//
#include <hip/hip_runtime.h>
#include <hip/hip_bf16.h>
#include <math.h>

// ---------------- problem constants ----------------
#define BB 2
#define TT 2048
#define DM 256
#define VV 32000
#define LL 4
#define DI 512
#define DS 16
#define DC 4
#define DTR 16
#define NTOK (BB*TT)   // 4096

typedef __attribute__((ext_vector_type(16))) __bf16 v16bf;
typedef __attribute__((ext_vector_type(4)))  __bf16 v4bf;
typedef __attribute__((ext_vector_type(8)))  float  v8f;

__device__ __forceinline__ v4bf cvt4_bf16(float4 v) {
    v4bf r;
    r.x = (__bf16)v.x; r.y = (__bf16)v.y; r.z = (__bf16)v.z; r.w = (__bf16)v.w;
    return r;
}

__device__ __forceinline__ unsigned short f32_to_bf16_rne(float f) {
    union { float f; unsigned u; } x; x.f = f;
    unsigned u = x.u + 0x7FFFu + ((x.u >> 16) & 1u);
    return (unsigned short)(u >> 16);
}

__device__ __forceinline__ float silu(float v) { return v / (1.f + __expf(-v)); }

// ============================================================
// bf16 WMMA GEMM:  O[M,N] = epilogue( A[M,K] @ W[N,K]^T )
// Block: 256 threads = 8 waves, macro tile 64(M) x 64(N), K step 32.
// Wave w: N-tile = w&3, M-tiles = {2*(w>>2), 2*(w>>2)+1}  (2 WMMAs/wave/step)
//
// Fast path: software-pipelined with double-buffered LDS:
//   iter i: [global_load_b128 x4 for slab i+1] -> [fragments + 2 WMMA from
//   buf p] -> [cvt_pk_bf16 + ds_store_b64 x4 into buf p^1] -> barrier.
// LDS tiles are pre-swizzled into the wave32 WMMA fragment layout:
//   half = (k>>3)&1 ; e = (k&7) | ((k>>4)<<3) ; lane = row%16 + 16*half
// so each lane reads a contiguous 32B v16bf fragment (2x ds_load_b128),
// and a 4-aligned k-quad is one contiguous 8B LDS store.
//
// EPI: 0 = none, 1 = +bias then softplus, 2 = accumulate into O (residual)
// GUARD: bounds-checked scalar staging (tiny ragged GEMMs only)
// ============================================================
template<int EPI, bool GUARD>
__global__ __launch_bounds__(256)
void gemm_bf16_wmma(const float* __restrict__ A, int lda,
                    const float* __restrict__ W, int ldw,
                    float* __restrict__ O, int ldo,
                    const float* __restrict__ bias,
                    int M, int N, int K)
{
    __shared__ __align__(16) unsigned short sA[2][4 * 32 * 16]; // 2 bufs x 4KB
    __shared__ __align__(16) unsigned short sW[2][4 * 32 * 16];

    const int tid   = threadIdx.x;
    const int lane  = tid & 31;
    const int wave  = tid >> 5;
    const int waveN = wave & 3;          // 0..3
    const int mt0   = (wave >> 2) << 1;  // 0 or 2
    const int m0 = blockIdx.y * 64;
    const int n0 = blockIdx.x * 64;

    v8f acc0 = {0.f,0.f,0.f,0.f,0.f,0.f,0.f,0.f};
    v8f acc1 = {0.f,0.f,0.f,0.f,0.f,0.f,0.f,0.f};

    const int fragA0 = (mt0       << 9) + (lane << 4);
    const int fragA1 = ((mt0 + 1) << 9) + (lane << 4);
    const int fragW  = (waveN     << 9) + (lane << 4);

    if (!GUARD) {
        // per-thread staging coordinates
        const int rL = tid >> 3;              // row 0..31 (and +32 sibling)
        const int kq = (tid & 7) << 2;        // k quad 0,4,...,28
        const int half = (kq >> 3) & 1;
        const int e0   = (kq & 7) | ((kq >> 4) << 3);
        const int off0 = ((rL >> 4) << 9) + ((((rL & 15) + (half << 4))) << 4) + e0;
        const int off1 = off0 + 1024;         // rows +32 -> tiles +2
        const float* aP = &A[(long)(m0 + rL) * lda + kq];
        const float* wP = &W[(long)(n0 + rL) * ldw + kq];
        const long aStep = (long)32 * lda;
        const long wStep = (long)32 * ldw;

        float4 a0, a1, w0, w1;
        // prologue: load + stage slab 0 into buf 0
        a0 = *(const float4*)(aP);
        a1 = *(const float4*)(aP + aStep);
        w0 = *(const float4*)(wP);
        w1 = *(const float4*)(wP + wStep);
        *(v4bf*)&sA[0][off0] = cvt4_bf16(a0);
        *(v4bf*)&sA[0][off1] = cvt4_bf16(a1);
        *(v4bf*)&sW[0][off0] = cvt4_bf16(w0);
        *(v4bf*)&sW[0][off1] = cvt4_bf16(w1);
        __syncthreads();

        int buf = 0;
        for (int k0 = 0; k0 < K; k0 += 32) {
            const int nk = k0 + 32;
            const bool more = nk < K;
            if (more) {                       // issue next-slab global loads early
                a0 = *(const float4*)(aP + nk);
                a1 = *(const float4*)(aP + aStep + nk);
                w0 = *(const float4*)(wP + nk);
                w1 = *(const float4*)(wP + wStep + nk);
            }
            // compute current slab from LDS
            v16bf af0 = *(const v16bf*)&sA[buf][fragA0];
            v16bf af1 = *(const v16bf*)&sA[buf][fragA1];
            v16bf wf  = *(const v16bf*)&sW[buf][fragW];
            acc0 = __builtin_amdgcn_wmma_f32_16x16x32_bf16(
                       false, af0, false, wf, (short)0, acc0, false, false);
            acc1 = __builtin_amdgcn_wmma_f32_16x16x32_bf16(
                       false, af1, false, wf, (short)0, acc1, false, false);
            if (more) {                       // stage next slab into other buffer
                *(v4bf*)&sA[buf ^ 1][off0] = cvt4_bf16(a0);
                *(v4bf*)&sA[buf ^ 1][off1] = cvt4_bf16(a1);
                *(v4bf*)&sW[buf ^ 1][off0] = cvt4_bf16(w0);
                *(v4bf*)&sW[buf ^ 1][off1] = cvt4_bf16(w1);
            }
            __syncthreads();
            buf ^= 1;
        }
    } else {
        // guarded scalar staging (tiny GEMMs only), two barriers per step
        for (int k0 = 0; k0 < K; k0 += 32) {
            #pragma unroll
            for (int i = 0; i < 8; ++i) {
                int idx = tid + i * 256;          // 0..2047
                int rL = idx >> 5, k = idx & 31;
                int gk = k0 + k;
                int half = (k >> 3) & 1;
                int e    = (k & 7) | ((k >> 4) << 3);
                int off  = ((rL >> 4) << 9) + ((((rL & 15) + (half << 4))) << 4) + e;
                float va = (m0 + rL < M && gk < K) ? A[(long)(m0 + rL) * lda + gk] : 0.f;
                float vw = (n0 + rL < N && gk < K) ? W[(long)(n0 + rL) * ldw + gk] : 0.f;
                sA[0][off] = f32_to_bf16_rne(va);
                sW[0][off] = f32_to_bf16_rne(vw);
            }
            __syncthreads();
            v16bf af0 = *(const v16bf*)&sA[0][fragA0];
            v16bf af1 = *(const v16bf*)&sA[0][fragA1];
            v16bf wf  = *(const v16bf*)&sW[0][fragW];
            acc0 = __builtin_amdgcn_wmma_f32_16x16x32_bf16(
                       false, af0, false, wf, (short)0, acc0, false, false);
            acc1 = __builtin_amdgcn_wmma_f32_16x16x32_bf16(
                       false, af1, false, wf, (short)0, acc1, false, false);
            __syncthreads();
        }
    }

    // epilogue: C/D layout -> lane n = lane%16, vgpr r -> m = r + 8*(lane/16)
    int n = n0 + (waveN << 4) + (lane & 15);
    if (!GUARD || n < N) {
        #pragma unroll
        for (int r = 0; r < 8; ++r) {
            int mrow = r + ((lane >> 4) << 3);
            #pragma unroll
            for (int q = 0; q < 2; ++q) {
                int m = m0 + ((mt0 + q) << 4) + mrow;
                if (!GUARD || m < M) {
                    float v = (q == 0) ? acc0[r] : acc1[r];
                    if (EPI == 1) {
                        v += bias[n];
                        v = (v > 20.f) ? v : log1pf(__expf(v));   // softplus
                    }
                    float* dst = &O[(long)m * ldo + n];
                    if (EPI == 2) v += *dst;                      // residual
                    *dst = v;
                }
            }
        }
    }
}

// ---------------- embedding: x = tok_emb[ids] + pos_emb ----------------
__global__ __launch_bounds__(256)
void embed_kernel(const int* __restrict__ ids, const float* __restrict__ tok,
                  const float* __restrict__ pos, float* __restrict__ x)
{
    int idx = blockIdx.x * 256 + threadIdx.x;
    if (idx >= NTOK * DM) return;
    int d  = idx % DM;
    int t  = (idx / DM) % TT;
    int bt = idx / DM;
    int id = ids[bt];
    x[idx] = tok[(long)id * DM + d] + pos[(long)t * DM + d];
}

// ---------------- causal depthwise conv (DC taps) + bias + silu ----------------
__global__ __launch_bounds__(256)
void conv_silu_kernel(const float* __restrict__ xz, const float* __restrict__ cw,
                      const float* __restrict__ cb, float* __restrict__ u)
{
    int idx = blockIdx.x * 256 + threadIdx.x;
    if (idx >= NTOK * DI) return;
    int c = idx % DI;
    int t = (idx / DI) % TT;
    long b = idx / (DI * TT);
    const float* base = xz + (b * TT) * (long)(2 * DI) + c;  // u = first DI cols
    float acc = cb[c];
    #pragma unroll
    for (int j = 0; j < DC; ++j) {
        int tt = t - (DC - 1) + j;
        if (tt >= 0) acc += cw[c * DC + j] * base[(long)tt * (2 * DI)];
    }
    u[idx] = silu(acc);
}

// ---------------- selective scan (sequential in T, parallel over B x DI) ------
__global__ __launch_bounds__(DI)
void scan_kernel(const float* __restrict__ delta,  // [B,T,DI] (softplus'd)
                 const float* __restrict__ u,      // [B,T,DI]
                 const float* __restrict__ xdbl,   // [B,T,48] (dt|B|C)
                 const float* __restrict__ xz,     // [B,T,2*DI] (z at +DI)
                 const float* __restrict__ A_log,  // [DI,DS] layer slice
                 const float* __restrict__ Dp,     // [DI]
                 float* __restrict__ y)            // [B,T,DI]
{
    int b = blockIdx.x;
    int d = threadIdx.x;
    __shared__ float sB[DS], sC[DS];

    float Av[DS];
    #pragma unroll
    for (int s = 0; s < DS; ++s) Av[s] = -__expf(A_log[d * DS + s]);
    float h[DS];
    #pragma unroll
    for (int s = 0; s < DS; ++s) h[s] = 0.f;
    float Dd = Dp[d];

    for (int t = 0; t < TT; ++t) {
        long row = (long)b * TT + t;
        __syncthreads();
        if (d < DS)            sB[d]      = xdbl[row * 48 + DTR + d];
        else if (d < 2 * DS)   sC[d - DS] = xdbl[row * 48 + DTR + DS + (d - DS)];
        __syncthreads();
        float dlt = delta[row * DI + d];
        float ut  = u[row * DI + d];
        float du  = dlt * ut;
        float acc = 0.f;
        #pragma unroll
        for (int s = 0; s < DS; ++s) {
            h[s] = __expf(dlt * Av[s]) * h[s] + du * sB[s];
            acc += h[s] * sC[s];
        }
        float zt = xz[row * (2 * DI) + DI + d];
        y[row * DI + d] = (acc + ut * Dd) * silu(zt);
    }
}

// ---------------- layernorm over DM=256 (one block per row) ----------------
__global__ __launch_bounds__(DM)
void layernorm_kernel(const float* __restrict__ x, const float* __restrict__ w,
                      const float* __restrict__ bias, float* __restrict__ o)
{
    int row = blockIdx.x, tid = threadIdx.x;
    __shared__ float red[DM];
    float v = x[(long)row * DM + tid];
    red[tid] = v; __syncthreads();
    for (int s = DM / 2; s > 0; s >>= 1) { if (tid < s) red[tid] += red[tid + s]; __syncthreads(); }
    float mu = red[0] / DM; __syncthreads();
    float dv = v - mu;
    red[tid] = dv * dv; __syncthreads();
    for (int s = DM / 2; s > 0; s >>= 1) { if (tid < s) red[tid] += red[tid + s]; __syncthreads(); }
    float inv = rsqrtf(red[0] / DM + 1e-5f);
    o[(long)row * DM + tid] = dv * inv * w[tid] + bias[tid];
}

// ---------------- per-row log-softmax NLL ----------------
__global__ __launch_bounds__(256)
void loss_row_kernel(const float* __restrict__ logits, const int* __restrict__ targets,
                     float* __restrict__ nll, float* __restrict__ cnt)
{
    int row = blockIdx.x, tid = threadIdx.x;
    __shared__ float red[256];
    const float* lr = logits + (long)row * VV;
    float mx = -INFINITY;
    for (int i = tid; i < VV; i += 256) mx = fmaxf(mx, lr[i]);
    red[tid] = mx; __syncthreads();
    for (int s = 128; s > 0; s >>= 1) { if (tid < s) red[tid] = fmaxf(red[tid], red[tid + s]); __syncthreads(); }
    mx = red[0]; __syncthreads();
    float sum = 0.f;
    for (int i = tid; i < VV; i += 256) sum += __expf(lr[i] - mx);
    red[tid] = sum; __syncthreads();
    for (int s = 128; s > 0; s >>= 1) { if (tid < s) red[tid] += red[tid + s]; __syncthreads(); }
    if (tid == 0) {
        int tg = targets[row];
        bool valid = (tg != -100);
        int t2 = valid ? tg : 0;
        float logp = lr[t2] - mx - __logf(red[0]);
        nll[row] = valid ? -logp : 0.f;
        cnt[row] = valid ? 1.f : 0.f;
    }
}

__global__ __launch_bounds__(256)
void loss_final_kernel(const float* __restrict__ nll, const float* __restrict__ cnt,
                       float* __restrict__ out)
{
    __shared__ float rs[256], rc[256];
    int tid = threadIdx.x;
    float s = 0.f, c = 0.f;
    for (int i = tid; i < NTOK; i += 256) { s += nll[i]; c += cnt[i]; }
    rs[tid] = s; rc[tid] = c; __syncthreads();
    for (int k = 128; k > 0; k >>= 1) {
        if (tid < k) { rs[tid] += rs[tid + k]; rc[tid] += rc[tid + k]; }
        __syncthreads();
    }
    if (tid == 0) out[0] = rs[0] / fmaxf(rc[0], 1.f);
}

// ============================================================
extern "C" void kernel_launch(void* const* d_in, const int* in_sizes, int n_in,
                              void* d_out, int out_size, void* d_ws, size_t ws_size,
                              hipStream_t stream)
{
    const int*   ids   = (const int*)d_in[0];
    const int*   tgts  = (const int*)d_in[1];
    const float* tok   = (const float*)d_in[2];
    const float* pos   = (const float*)d_in[3];
    const float* ipw   = (const float*)d_in[4];   // [L, 2*DI, DM]
    const float* cw    = (const float*)d_in[5];   // [L, DI, DC]
    const float* cb    = (const float*)d_in[6];   // [L, DI]
    const float* xpw   = (const float*)d_in[7];   // [L, 48, DI]
    const float* dpw   = (const float*)d_in[8];   // [L, DI, DTR]
    const float* dpb   = (const float*)d_in[9];   // [L, DI]
    const float* Alog  = (const float*)d_in[10];  // [L, DI, DS]
    const float* Dpar  = (const float*)d_in[11];  // [L, DI]
    const float* opw   = (const float*)d_in[12];  // [L, DM, DI]
    const float* lnw   = (const float*)d_in[13];
    const float* lnb   = (const float*)d_in[14];
    const float* hw    = (const float*)d_in[15];  // [V, DM]

    float* logits = (float*)d_out;                // [B,T,V]
    float* loss   = logits + (long)NTOK * VV;     // scalar

    // workspace layout (floats)
    float* ws = (float*)d_ws;
    float* x     = ws;                 ws += (long)NTOK * DM;
    float* xz    = ws;                 ws += (long)NTOK * 2 * DI;
    float* u     = ws;                 ws += (long)NTOK * DI;
    float* xdbl  = ws;                 ws += (long)NTOK * 48;
    float* delta = ws;                 ws += (long)NTOK * DI;
    float* y     = ws;                 ws += (long)NTOK * DI;
    float* hln   = ws;                 ws += (long)NTOK * DM;
    float* nll   = ws;                 ws += NTOK;
    float* cnt   = ws;                 ws += NTOK;

    // 1) embedding
    embed_kernel<<<(NTOK * DM + 255) / 256, 256, 0, stream>>>(ids, tok, pos, x);

    // 2) mamba layers
    for (int l = 0; l < LL; ++l) {
        const float* ipw_l = ipw + (long)l * 2 * DI * DM;
        const float* cw_l  = cw  + (long)l * DI * DC;
        const float* cb_l  = cb  + (long)l * DI;
        const float* xpw_l = xpw + (long)l * 48 * DI;
        const float* dpw_l = dpw + (long)l * DI * DTR;
        const float* dpb_l = dpb + (long)l * DI;
        const float* Al_l  = Alog + (long)l * DI * DS;
        const float* Dp_l  = Dpar + (long)l * DI;
        const float* opw_l = opw + (long)l * DM * DI;

        // xz = x @ ipw^T : (4096,256)x(1024,256)^T  [aligned -> fast path]
        gemm_bf16_wmma<0, false><<<dim3(2 * DI / 64, NTOK / 64), 256, 0, stream>>>(
            x, DM, ipw_l, DM, xz, 2 * DI, nullptr, NTOK, 2 * DI, DM);

        // u = silu(causal_conv(xz[:, :DI]) + cb)
        conv_silu_kernel<<<(NTOK * DI + 255) / 256, 256, 0, stream>>>(xz, cw_l, cb_l, u);

        // x_dbl = u @ xpw^T : (4096,512)x(48,512)^T  [N=48 ragged -> guarded]
        gemm_bf16_wmma<0, true><<<dim3(1, NTOK / 64), 256, 0, stream>>>(
            u, DI, xpw_l, DI, xdbl, 48, nullptr, NTOK, 48, DI);

        // delta = softplus(x_dbl[:, :16] @ dpw^T + dpb) : K=16 ragged -> guarded
        gemm_bf16_wmma<1, true><<<dim3(DI / 64, NTOK / 64), 256, 0, stream>>>(
            xdbl, 48, dpw_l, DTR, delta, DI, dpb_l, NTOK, DI, DTR);

        // selective scan + D skip + silu(z) gate
        scan_kernel<<<BB, DI, 0, stream>>>(delta, u, xdbl, xz, Al_l, Dp_l, y);

        // x += y @ opw^T : (4096,512)x(256,512)^T, residual epilogue [fast]
        gemm_bf16_wmma<2, false><<<dim3(DM / 64, NTOK / 64), 256, 0, stream>>>(
            y, DI, opw_l, DI, x, DM, nullptr, NTOK, DM, DI);
    }

    // 3) layernorm
    layernorm_kernel<<<NTOK, DM, 0, stream>>>(x, lnw, lnb, hln);

    // 4) logits = hln @ head_w^T : (4096,256)x(32000,256)^T  [fast]
    gemm_bf16_wmma<0, false><<<dim3(VV / 64, NTOK / 64), 256, 0, stream>>>(
        hln, DM, hw, DM, logits, VV, nullptr, NTOK, VV, DM);

    // 5) loss (deterministic two-pass)
    loss_row_kernel<<<NTOK, 256, 0, stream>>>(logits, tgts, nll, cnt);
    loss_final_kernel<<<1, 256, 0, stream>>>(nll, cnt, loss);
}
